// DecoderLayer_45629732552916
// MI455X (gfx1250) — matrix operand order repaired and checked by reading, not compiled, not run on hardware
//
#include <hip/hip_runtime.h>
#include <hip/hip_bf16.h>

typedef __attribute__((ext_vector_type(16))) _Float16 v16h;
typedef __attribute__((ext_vector_type(8)))  _Float16 v8h;
typedef __attribute__((ext_vector_type(8)))  float    v8f;

// ---------------------------------------------------------------------------
// CDNA5-specific helpers
// ---------------------------------------------------------------------------
__device__ __forceinline__ v8f wmma_f32(v16h a, v16h b, v8f c) {
    // 8 args: (neg_a, A, neg_b, B, c_mod, C, reuse_a, reuse_b)
    return __builtin_amdgcn_wmma_f32_16x16x32_f16(false, a, false, b,
                                                  (short)0, c, false, false);
}

// Async global->LDS copy (8 bytes per lane, GV addressing, ASYNCcnt-tracked).
__device__ __forceinline__ void async_b64(unsigned lds_off, const void* g) {
    asm volatile("global_load_async_to_lds_b64 %0, %1, off"
                 :: "v"(lds_off), "v"(g) : "memory");
}
__device__ __forceinline__ void wait_async() {
    asm volatile("s_wait_asynccnt 0x0" ::: "memory");
}
__device__ __forceinline__ unsigned lds_addr32(const void* p) {
    // LDS aperture keeps the LDS byte offset in the low 32 bits of the flat addr.
    return (unsigned)(unsigned long long)p;
}

// A-matrix fragment (16x32 f16), ISA 7.12.2: lane kg = lane>>4 holds
// K = [8*kg,8*kg+8) in elems 0..7 and K = [16+8*kg,24+8*kg) in elems 8..15.
__device__ __forceinline__ v16h ld_frag_a(const _Float16* rowbase, int kg) {
    v8h x0 = *(const v8h*)(rowbase + 8 * kg);
    v8h x1 = *(const v8h*)(rowbase + 16 + 8 * kg);
    v16h r;
#pragma unroll
    for (int e = 0; e < 8; ++e) { r[e] = x0[e]; r[e + 8] = x1[e]; }
    return r;
}

// B-matrix fragment (32x16 f16): lane holds column N = lane&15, elems 0..7 map to
// K = 8*kg+e, elems 8..15 to K = 16+8*kg+(e-8). colbase = this lane's column in TRANSPOSED [n][k] tile.
__device__ __forceinline__ v16h ld_frag_b(const _Float16* colbase, int kg) {
    v8h x0 = *(const v8h*)(colbase + 8 * kg);
    v8h x1 = *(const v8h*)(colbase + 16 + 8 * kg);
    v16h r;
#pragma unroll
    for (int e = 0; e < 8; ++e) { r[e] = x0[e]; r[e + 8] = x1[e]; }
    return r;
}

// ---------------------------------------------------------------------------
// GEMM: C[M,N] = act(A[M,K] @ W[K,N] + bias)
// A: f16 (async-staged to LDS), W: f32 (converted while staging), C: f16 out.
// Block = 128 threads (4 waves), tile 64x64, K-step 32; 2x2 WMMA per wave.
// ---------------------------------------------------------------------------
template <bool RELU>
__global__ __launch_bounds__(128) void gemm_kernel(
    const _Float16* __restrict__ A, const float* __restrict__ W,
    const float* __restrict__ bias, _Float16* __restrict__ C,
    int M, int N, int K)
{
    constexpr int BM = 64, BN = 64, BK = 32;
    constexpr int LDA = BK + 8;   // 40 halves: 80B rows, 16B aligned
    constexpr int LDB = BK + 8;
    __shared__ __align__(16) _Float16 sA[BM * LDA];
    __shared__ __align__(16) _Float16 sB[BN * LDB];   // transposed [n][k]

    const int tid  = threadIdx.x;
    const int lane = tid & 31, wave = tid >> 5;
    const int kg = lane >> 4, l16 = lane & 15;
    const int m_off = (wave >> 1) * 32, n_off = (wave & 1) * 32;
    const int gm = blockIdx.x * BM, gn = blockIdx.y * BN;
    const unsigned ldsA = lds_addr32(&sA[0]);

    v8f acc[2][2] = {};

    for (int k0 = 0; k0 < K; k0 += BK) {
        if (k0 + BK < K && tid < BK)   // prefetch next W tile rows
            __builtin_prefetch(&W[(size_t)(k0 + BK + tid) * N + gn], 0, 1);

        // A tile (64 rows x 32 halves): async global->LDS, 8B per lane
        const _Float16* Ab = A + (size_t)gm * K + k0;
        for (int c = tid; c < BM * (BK / 4); c += 128) {
            int r = c >> 3, cc = c & 7;
            async_b64(ldsA + (unsigned)(r * LDA + cc * 4) * 2,
                      Ab + (size_t)r * K + cc * 4);
        }
        // W tile transposed + f32->f16: sB[n][k]
        for (int i = tid; i < BK * BN; i += 128) {
            int kk = i >> 6, n = i & 63;
            sB[n * LDB + kk] = (_Float16)W[(size_t)(k0 + kk) * N + (gn + n)];
        }
        wait_async();
        __syncthreads();

        v16h a0 = ld_frag_a(&sA[(m_off + l16) * LDA], kg);
        v16h a1 = ld_frag_a(&sA[(m_off + 16 + l16) * LDA], kg);
        v16h b0 = ld_frag_b(&sB[(n_off + l16) * LDB], kg);
        v16h b1 = ld_frag_b(&sB[(n_off + 16 + l16) * LDB], kg);
        acc[0][0] = wmma_f32(a0, b0, acc[0][0]);
        acc[0][1] = wmma_f32(a0, b1, acc[0][1]);
        acc[1][0] = wmma_f32(a1, b0, acc[1][0]);
        acc[1][1] = wmma_f32(a1, b1, acc[1][1]);
        __syncthreads();
    }

#pragma unroll
    for (int mi = 0; mi < 2; ++mi)
#pragma unroll
        for (int ni = 0; ni < 2; ++ni)
#pragma unroll
            for (int i = 0; i < 8; ++i) {
                int row = gm + m_off + mi * 16 + i + 8 * kg;
                int col = gn + n_off + ni * 16 + l16;
                float v = acc[mi][ni][i] + bias[col];
                if (RELU) v = fmaxf(v, 0.0f);
                C[(size_t)row * N + col] = (_Float16)v;
            }
}

// ---------------------------------------------------------------------------
// Attention: one block per (sequence, head, 16-row q-tile). f16 in/out.
// scores = Q @ K^T / sqrt(32) (+causal), softmax(f32), O = P @ V.
// ---------------------------------------------------------------------------
template <int TK, bool CAUSAL>
__global__ __launch_bounds__(128) void attn_kernel(
    const _Float16* __restrict__ Qg, const _Float16* __restrict__ Kg,
    const _Float16* __restrict__ Vg, _Float16* __restrict__ Og, int Tq)
{
    constexpr int HD = 32;
    constexpr int LQK = HD + 8;    // 40 halves
    constexpr int LVP = TK + 8;    // 104/200 halves -> 16B-aligned rows
    constexpr int LS  = TK + 4;    // floats
    __shared__ __align__(16) _Float16 sQ[16 * LQK];
    __shared__ __align__(16) _Float16 sK[TK * LQK];
    __shared__ __align__(16) _Float16 sVT[HD * LVP];  // transposed [hd][key]
    __shared__ __align__(16) float    sS[16 * LS];
    __shared__ __align__(16) _Float16 sP[16 * LVP];

    const int tid  = threadIdx.x;
    const int lane = tid & 31, wave = tid >> 5;
    const int kg = lane >> 4, l16 = lane & 15;

    const int nqt = Tq / 16;
    const int qt  = blockIdx.x % nqt;
    const int sh  = blockIdx.x / nqt;
    const int h   = sh % 8, s = sh / 8;

    const size_t qbase = ((size_t)s * Tq + qt * 16) * 256 + h * HD;
    const size_t kbase = (size_t)s * TK * 256 + h * HD;

    // Q and K tiles: async global->LDS (f16, no conversion needed)
    const unsigned ldsQ = lds_addr32(&sQ[0]);
    const unsigned ldsK = lds_addr32(&sK[0]);
    for (int c = tid; c < 16 * (HD / 4); c += 128) {
        int r = c >> 3, cc = c & 7;
        async_b64(ldsQ + (unsigned)(r * LQK + cc * 4) * 2,
                  Qg + qbase + (size_t)r * 256 + cc * 4);
    }
    for (int c = tid; c < TK * (HD / 4); c += 128) {
        int r = c >> 3, cc = c & 7;
        async_b64(ldsK + (unsigned)(r * LQK + cc * 4) * 2,
                  Kg + kbase + (size_t)r * 256 + cc * 4);
    }
    // V: regular loads, transposed LDS writes
    for (int i = tid; i < TK * HD; i += 128) {
        int r = i >> 5, c = i & 31;
        sVT[c * LVP + r] = Vg[kbase + (size_t)r * 256 + c];
    }
    wait_async();
    __syncthreads();

    const float scale = 0.17677669529663687f;  // 1/sqrt(32)
    v16h aq = ld_frag_a(&sQ[l16 * LQK], kg);
    for (int ct = wave; ct < TK / 16; ct += 4) {
        v16h bk = ld_frag_b(&sK[(ct * 16 + l16) * LQK], kg);  // K rows = K^T cols
        v8f c = {};
        c = wmma_f32(aq, bk, c);
#pragma unroll
        for (int i = 0; i < 8; ++i) {
            int r   = i + 8 * kg;        // q row within tile
            int col = ct * 16 + l16;     // key index
            float v = c[i] * scale;
            if (CAUSAL && col > qt * 16 + r) v = -__builtin_inff();
            sS[r * LS + col] = v;
        }
    }
    __syncthreads();

    if (tid < 16) {  // one thread per q row
        float mx = -__builtin_inff();
        for (int c = 0; c < TK; ++c) mx = fmaxf(mx, sS[tid * LS + c]);
        float sum = 0.0f;
        for (int c = 0; c < TK; ++c) {
            float e = __expf(sS[tid * LS + c] - mx);
            sum += e;
            sS[tid * LS + c] = e;
        }
        float inv = 1.0f / sum;
        for (int c = 0; c < TK; ++c)
            sP[tid * LVP + c] = (_Float16)(sS[tid * LS + c] * inv);
    }
    __syncthreads();

    if (wave < 2) {  // O = P[16xTK] @ V[TKx32]; wave picks 16-col half
        v8f acc = {};
        for (int kb = 0; kb < TK; kb += 32) {
            v16h ap = ld_frag_a(&sP[l16 * LVP + kb], kg);
            v16h bv = ld_frag_b(&sVT[(wave * 16 + l16) * LVP + kb], kg);
            acc = wmma_f32(ap, bv, acc);
        }
#pragma unroll
        for (int i = 0; i < 8; ++i)
            Og[qbase + (size_t)(i + 8 * kg) * 256 + wave * 16 + l16] =
                (_Float16)acc[i];
    }
}

// ---------------------------------------------------------------------------
// LayerNorm over last dim (256): out = LN(A + B) * g + beta. One row / block.
// Reduction in fp32 regardless of input/output types.
// ---------------------------------------------------------------------------
template <typename TA, typename TB, typename TOUT>
__global__ __launch_bounds__(256) void ln_kernel(
    const TA* __restrict__ Ain, const TB* __restrict__ Bin,
    const float* __restrict__ g, const float* __restrict__ beta,
    TOUT* __restrict__ out)
{
    __shared__ float red[256];
    const int row = blockIdx.x, t = threadIdx.x;
    float v = (float)Ain[(size_t)row * 256 + t] + (float)Bin[(size_t)row * 256 + t];
    red[t] = v; __syncthreads();
    for (int s = 128; s > 0; s >>= 1) { if (t < s) red[t] += red[t + s]; __syncthreads(); }
    float mean = red[0] * (1.0f / 256.0f);
    __syncthreads();
    float d = v - mean;
    red[t] = d * d; __syncthreads();
    for (int s = 128; s > 0; s >>= 1) { if (t < s) red[t] += red[t + s]; __syncthreads(); }
    float var = red[0] * (1.0f / 256.0f);
    float rs = rsqrtf(var + 1e-5f);
    out[(size_t)row * 256 + t] = (TOUT)(d * rs * g[t] + beta[t]);
}

// ---------------------------------------------------------------------------
// Transpose: in [B][X][Y][256] -> out [B][Y][X][256]. One 256-elem row / block.
// ---------------------------------------------------------------------------
__global__ __launch_bounds__(256) void transpose_cast_kernel(
    const float* __restrict__ in, _Float16* __restrict__ out, int X, int Y)
{
    const int t = threadIdx.x;
    const size_t bi = blockIdx.x;
    const int y  = (int)(bi % Y);
    const int xr = (int)((bi / Y) % X);
    const int b  = (int)(bi / ((size_t)Y * X));
    out[(((size_t)b * Y + y) * X + xr) * 256 + t] = (_Float16)in[bi * 256 + t];
}

__global__ __launch_bounds__(256) void transpose_h_kernel(
    const _Float16* __restrict__ in, _Float16* __restrict__ out, int X, int Y)
{
    const int t = threadIdx.x;
    const size_t bi = blockIdx.x;
    const int y  = (int)(bi % Y);
    const int xr = (int)((bi / Y) % X);
    const int b  = (int)(bi / ((size_t)Y * X));
    out[(((size_t)b * Y + y) * X + xr) * 256 + t] = in[bi * 256 + t];
}

// ---------------------------------------------------------------------------
extern "C" void kernel_launch(void* const* d_in, const int* in_sizes, int n_in,
                              void* d_out, int out_size, void* d_ws, size_t ws_size,
                              hipStream_t stream)
{
    (void)in_sizes; (void)n_in; (void)out_size; (void)ws_size;
    const float* x        = (const float*)d_in[0];
    const float* memory   = (const float*)d_in[1];
    const float* sa_wq    = (const float*)d_in[2];
    const float* sa_bq    = (const float*)d_in[3];
    const float* sa_wk    = (const float*)d_in[4];
    const float* sa_bk    = (const float*)d_in[5];
    const float* sa_wv    = (const float*)d_in[6];
    const float* sa_bv    = (const float*)d_in[7];
    const float* sa_wo    = (const float*)d_in[8];
    const float* sa_bo    = (const float*)d_in[9];
    const float* sa_ff_w1 = (const float*)d_in[10];
    const float* sa_ff_b1 = (const float*)d_in[11];
    const float* sa_ff_w2 = (const float*)d_in[12];
    const float* sa_ff_b2 = (const float*)d_in[13];
    const float* sa_ln1_g = (const float*)d_in[14];
    const float* sa_ln1_b = (const float*)d_in[15];
    const float* sa_ln2_g = (const float*)d_in[16];
    const float* sa_ln2_b = (const float*)d_in[17];
    const float* ca_wq    = (const float*)d_in[18];
    const float* ca_bq    = (const float*)d_in[19];
    const float* ca_wk    = (const float*)d_in[20];
    const float* ca_bk    = (const float*)d_in[21];
    const float* ca_wv    = (const float*)d_in[22];
    const float* ca_bv    = (const float*)d_in[23];
    const float* ca_wo    = (const float*)d_in[24];
    const float* ca_bo    = (const float*)d_in[25];
    const float* ff_w1    = (const float*)d_in[26];
    const float* ff_b1    = (const float*)d_in[27];
    const float* ff_w2    = (const float*)d_in[28];
    const float* ff_b2    = (const float*)d_in[29];
    const float* ln1_g    = (const float*)d_in[30];
    const float* ln1_b    = (const float*)d_in[31];
    const float* ln2_g    = (const float*)d_in[32];
    const float* ln2_b    = (const float*)d_in[33];
    const float* ln3_g    = (const float*)d_in[34];
    const float* ln3_b    = (const float*)d_in[35];

    const int S = 8 * 64;                 // B * HW = 512 sequences
    const int M1 = S * 96;                // 49152 query tokens
    const int M2 = S * 192;               // 98304 memory tokens
    const size_t E1 = (size_t)M1 * 256;
    const size_t E2 = (size_t)M2 * 256;
    const size_t EF = (size_t)M1 * 1024;

    _Float16* ws = (_Float16*)d_ws;       // all activations kept in f16
    _Float16* hA = ws;
    _Float16* hB = hA + E1;
    _Float16* hC = hB + E1;
    _Float16* hD = hC + E1;
    _Float16* hE = hD + E1;
    _Float16* hF = hE + E1;               // FFN hidden (M1 x 1024)
    _Float16* hK = hF + EF;               // cross K (M2 x 256)
    _Float16* hV = hK + E2;               // cross V (M2 x 256)
    _Float16* hM = hF;                    // mem^T aliases hF (disjoint lifetime)

    const dim3 blk128(128), blk256(256);
    const dim3 gD(M1 / 64, 256 / 64);     // (768, 4)
    const dim3 gF(M1 / 64, 1024 / 64);    // (768, 16)
    const dim3 gM(M2 / 64, 256 / 64);     // (1536, 4)
    const int attn_blocks = S * 8 * (96 / 16);   // 24576

    // ---- temporal self-attention (t-major: [B][HW][T][D]) ----
    transpose_cast_kernel<<<8 * 96 * 64, blk256, 0, stream>>>(x, hA, 96, 64);
    gemm_kernel<false><<<gD, blk128, 0, stream>>>(hA, sa_wq, sa_bq, hB, M1, 256, 256);
    gemm_kernel<false><<<gD, blk128, 0, stream>>>(hA, sa_wk, sa_bk, hC, M1, 256, 256);
    gemm_kernel<false><<<gD, blk128, 0, stream>>>(hA, sa_wv, sa_bv, hD, M1, 256, 256);
    attn_kernel<96, true><<<attn_blocks, blk128, 0, stream>>>(hB, hC, hD, hE, 96);
    gemm_kernel<false><<<gD, blk128, 0, stream>>>(hE, sa_wo, sa_bo, hB, M1, 256, 256);
    ln_kernel<<<M1, blk256, 0, stream>>>(hA, hB, sa_ln1_g, sa_ln1_b, hC);
    gemm_kernel<true ><<<gF, blk128, 0, stream>>>(hC, sa_ff_w1, sa_ff_b1, hF, M1, 1024, 256);
    gemm_kernel<false><<<gD, blk128, 0, stream>>>(hF, sa_ff_w2, sa_ff_b2, hD, M1, 256, 1024);
    ln_kernel<<<M1, blk256, 0, stream>>>(hC, hD, sa_ln2_g, sa_ln2_b, hB);
    transpose_h_kernel<<<8 * 96 * 64, blk256, 0, stream>>>(hB, hA, 64, 96);  // -> [B][T][HW][D]
    ln_kernel<<<M1, blk256, 0, stream>>>(hA, x, ln1_g, ln1_b, hC);           // x1 (f32 residual x)

    // ---- cross-attention over time ----
    transpose_h_kernel<<<8 * 96 * 64, blk256, 0, stream>>>(hC, hA, 96, 64);  // x1 t-major
    gemm_kernel<false><<<gD, blk128, 0, stream>>>(hA, ca_wq, ca_bq, hB, M1, 256, 256);
    transpose_cast_kernel<<<8 * 192 * 64, blk256, 0, stream>>>(memory, hM, 192, 64);
    gemm_kernel<false><<<gM, blk128, 0, stream>>>(hM, ca_wk, ca_bk, hK, M2, 256, 256);
    gemm_kernel<false><<<gM, blk128, 0, stream>>>(hM, ca_wv, ca_bv, hV, M2, 256, 256);
    attn_kernel<192, false><<<attn_blocks, blk128, 0, stream>>>(hB, hK, hV, hD, 96);
    gemm_kernel<false><<<gD, blk128, 0, stream>>>(hD, ca_wo, ca_bo, hE, M1, 256, 256);
    transpose_h_kernel<<<8 * 96 * 64, blk256, 0, stream>>>(hE, hB, 64, 96);  // co -> [B][T][HW][D]
    ln_kernel<<<M1, blk256, 0, stream>>>(hB, hC, ln2_g, ln2_b, hD);          // x2

    // ---- decoder FFN ----
    gemm_kernel<true ><<<gF, blk128, 0, stream>>>(hD, ff_w1, ff_b1, hF, M1, 1024, 256);
    gemm_kernel<false><<<gD, blk128, 0, stream>>>(hF, ff_w2, ff_b2, hE, M1, 256, 1024);
    ln_kernel<<<M1, blk256, 0, stream>>>(hD, hE, ln3_g, ln3_b, (float*)d_out);
}
